// LocalFeatureAggregation_15925738734009
// MI455X (gfx1250) — matrix operand order repaired and checked by reading, not compiled, not run on hardware
//
#include <hip/hip_runtime.h>
#include <hip/hip_fp16.h>

typedef __attribute__((ext_vector_type(16))) _Float16 v16h;
typedef __attribute__((ext_vector_type(8)))  float    v8f;

constexpr int NPTS = 8192;
constexpr int NBAT = 2;
constexpr int KNN  = 16;
constexpr int MN   = NBAT * NPTS;      // 16384 points
constexpr int MNK  = MN * KNN;         // 262144 point-neighbor pairs

// ---------------------------------------------------------------------------
// KNN: each thread owns one query, scans all support points via LDS tiles,
// keeps a sorted top-16 (ascending d2) in registers.
// ---------------------------------------------------------------------------
__global__ void knn_kernel(const float* __restrict__ xyz,
                           int* __restrict__ nbrs, float* __restrict__ dist) {
  __shared__ float sx[256], sy[256], sz[256];
  const int b = blockIdx.y;
  const int q = blockIdx.x * 256 + threadIdx.x;
  const float qx = xyz[(size_t)(b * NPTS + q) * 3 + 0];
  const float qy = xyz[(size_t)(b * NPTS + q) * 3 + 1];
  const float qz = xyz[(size_t)(b * NPTS + q) * 3 + 2];
  float bd[KNN]; int bi[KNN];
#pragma unroll
  for (int j = 0; j < KNN; ++j) { bd[j] = 3.0e38f; bi[j] = 0; }
  for (int t0 = 0; t0 < NPTS; t0 += 256) {
    __syncthreads();
    const int s = t0 + threadIdx.x;
    sx[threadIdx.x] = xyz[(size_t)(b * NPTS + s) * 3 + 0];
    sy[threadIdx.x] = xyz[(size_t)(b * NPTS + s) * 3 + 1];
    sz[threadIdx.x] = xyz[(size_t)(b * NPTS + s) * 3 + 2];
    __syncthreads();
    for (int j = 0; j < 256; ++j) {
      const float dx = qx - sx[j], dy = qy - sy[j], dz = qz - sz[j];
      const float d2 = dx * dx + dy * dy + dz * dz;
      if (d2 < bd[KNN - 1]) {
        bd[KNN - 1] = d2; bi[KNN - 1] = t0 + j;
#pragma unroll
        for (int u = KNN - 1; u > 0; --u) {
          if (bd[u] < bd[u - 1]) {
            float td = bd[u]; bd[u] = bd[u - 1]; bd[u - 1] = td;
            int   ti = bi[u]; bi[u] = bi[u - 1]; bi[u - 1] = ti;
          }
        }
      }
    }
  }
  const size_t base = (size_t)(b * NPTS + q) * KNN;
#pragma unroll
  for (int j = 0; j < KNN; ++j) {
    nbrs[base + j] = bi[j];
    dist[base + j] = sqrtf(fmaxf(bd[j], 1e-12f));
  }
}

// ---------------------------------------------------------------------------
// WMMA GEMM: Y(f32, Cout x M) = W(f16, Cout x CINPAD) * X(f16, M x ldX) + bias
// Each wave computes NCO=4 Cout-tiles (64 channels) x 16 points, reusing the
// B (activation) fragment across the 4 A (weight) fragments per K-step.
// All fragment loads are base-pointer + compile-time-constant offsets so the
// compiler can clause them and issue the 4 WMMAs back-to-back.
// bias must be non-null (use a zero vector for bias-free GEMMs).
// A/B per-lane packing per CDNA5 ISA 7.12.2 16-bit layouts.
// ---------------------------------------------------------------------------
template <int CINPAD, int NCO>
__global__ void gemm_wmma_f16(const _Float16* __restrict__ W,
                              const _Float16* __restrict__ X,
                              float* __restrict__ Y,
                              const float* __restrict__ bias,
                              int M, int ldX, int tiles) {
  const int wave = threadIdx.x >> 5;
  const int lane = threadIdx.x & 31;
  const int t = blockIdx.x * 8 + wave;
  if (t >= tiles) return;
  const int tilesM = M >> 4;
  const int m0  = (t % tilesM) << 4;
  const int cg0 = (t / tilesM) * (NCO * 16);
  const int half = lane >> 4;
  const int l16  = lane & 15;

  // Base pointers with all runtime components folded in; every fragment load
  // below is base + constexpr offset (fits 24-bit immediates).
  const _Float16* xb = X + (size_t)(m0 + l16) * ldX + 8 * half;       // B
  const _Float16* wb = W + (size_t)(cg0 + l16) * CINPAD + 8 * half;   // A

  const v8f vzero = {};
  v8f acc[NCO];
#pragma unroll
  for (int i = 0; i < NCO; ++i) acc[i] = vzero;

#pragma unroll
  for (int ci0 = 0; ci0 < CINPAD; ci0 += 32) {
    // B (32x16): lane col = l16; K offset (v<4 ? 2v : 2v+8) + 8*half
    v16h bm;
#pragma unroll
    for (int v = 0; v < 8; ++v) {
      const int kb = (v < 4) ? (2 * v) : (2 * v + 8);
      bm[2 * v]     = xb[ci0 + kb];
      bm[2 * v + 1] = xb[ci0 + kb + 1];
    }
    // A (16x32) x NCO tiles: lane row = l16; K offset (v<4 ? 2v : 2v+8) + 8*half
    v16h a[NCO];
#pragma unroll
    for (int i = 0; i < NCO; ++i) {
#pragma unroll
      for (int v = 0; v < 8; ++v) {
        const int ka = (v < 4) ? (2 * v) : (2 * v + 8);
        a[i][2 * v]     = wb[i * 16 * CINPAD + ci0 + ka];
        a[i][2 * v + 1] = wb[i * 16 * CINPAD + ci0 + ka + 1];
      }
    }
#pragma unroll
    for (int i = 0; i < NCO; ++i)
      acc[i] = __builtin_amdgcn_wmma_f32_16x16x32_f16(false, a[i], false, bm,
                                                      (short)0, acc[i], false, false);
  }

#pragma unroll
  for (int i = 0; i < NCO; ++i) {
    // 8 contiguous bias floats for this lane's rows: vectorizes to b128 loads
    const float* bp = bias + cg0 + i * 16 + 8 * half;
    float bv[8];
#pragma unroll
    for (int r = 0; r < 8; ++r) bv[r] = bp[r];
#pragma unroll
    for (int r = 0; r < 8; ++r) {
      const int co = cg0 + i * 16 + r + 8 * half;   // D row = r + 8*half
      Y[(size_t)co * M + m0 + l16] = acc[i][r] + bv[r];
    }
  }
}

// ---------------------------------------------------------------------------
// Per-channel training-mode BN stats (biased variance) over M samples.
// ---------------------------------------------------------------------------
__global__ void stats_kernel(const float* __restrict__ Y, int M,
                             float* __restrict__ mean, float* __restrict__ rstd) {
  __shared__ float sh1[256], sh2[256];
  const int c = blockIdx.x;
  const float* row = Y + (size_t)c * M;
  float s = 0.f, s2 = 0.f;
  for (int i = threadIdx.x; i < M; i += 256) { float v = row[i]; s += v; s2 += v * v; }
  sh1[threadIdx.x] = s; sh2[threadIdx.x] = s2;
  __syncthreads();
  for (int st = 128; st > 0; st >>= 1) {
    if ((int)threadIdx.x < st) {
      sh1[threadIdx.x] += sh1[threadIdx.x + st];
      sh2[threadIdx.x] += sh2[threadIdx.x + st];
    }
    __syncthreads();
  }
  if (threadIdx.x == 0) {
    const float mm = sh1[0] / (float)M;
    const float vv = sh2[0] / (float)M - mm * mm;
    mean[c] = mm;
    rstd[c] = rsqrtf(vv + 1e-6f);
  }
}

// BN + activation, write f16 into point-major dst at column offset (free concat).
// act: 0=none, 1=relu, 2=leaky(0.2)
__global__ void bn_act_f16_kernel(const float* __restrict__ Y,
                                  const float* __restrict__ mean,
                                  const float* __restrict__ rstd,
                                  const float* __restrict__ g,
                                  const float* __restrict__ be,
                                  _Float16* __restrict__ dst,
                                  int M, int ldD, int rowoff, int act, int total) {
  const int tid = blockIdx.x * 256 + threadIdx.x;
  if (tid >= total) return;
  const int c = tid / M, m = tid % M;
  float v = (Y[(size_t)c * M + m] - mean[c]) * rstd[c] * g[c] + be[c];
  if (act == 1) v = fmaxf(v, 0.f);
  else if (act == 2) v = (v > 0.f) ? v : 0.2f * v;
  dst[(size_t)m * ldD + rowoff + c] = (_Float16)v;
}

// BN in-place on f32 (no activation) for the two residual branches.
__global__ void bn_f32_kernel(float* __restrict__ Y,
                              const float* __restrict__ mean,
                              const float* __restrict__ rstd,
                              const float* __restrict__ g,
                              const float* __restrict__ be,
                              int M, int total) {
  const int tid = blockIdx.x * 256 + threadIdx.x;
  if (tid >= total) return;
  const int c = tid / M;
  Y[tid] = (Y[tid] - mean[c]) * rstd[c] * g[c] + be[c];
}

// feat_in (B,64,N,1) f32 -> point-major f16 (MN, 64)
__global__ void cvt_featin_kernel(const float* __restrict__ f, _Float16* __restrict__ dst) {
  const int tid = blockIdx.x * 256 + threadIdx.x;
  if (tid >= NBAT * 64 * NPTS) return;
  const int n = tid % NPTS, c = (tid / NPTS) % 64, b = tid / (64 * NPTS);
  dst[(size_t)(b * NPTS + n) * 64 + c] = (_Float16)f[tid];
}

// f32 weight (Cout,Cin) -> f16 (Cout,CinPad) with zero pad
__global__ void cvt_w_kernel(const float* __restrict__ src, _Float16* __restrict__ dst,
                             int Cin, int CinPad, int total) {
  const int tid = blockIdx.x * 256 + threadIdx.x;
  if (tid >= total) return;
  const int c = tid / CinPad, k = tid % CinPad;
  dst[tid] = (k < Cin) ? (_Float16)src[c * Cin + k] : (_Float16)0.0f;
}

// zero a small f32 buffer (used as the null bias vector)
__global__ void zero_f32_kernel(float* __restrict__ p, int n) {
  const int tid = blockIdx.x * 256 + threadIdx.x;
  if (tid < n) p[tid] = 0.f;
}

// Build 10-channel relative position encoding, point-major (MNK, 32), rows 10..31 = 0
__global__ void rpe_build_kernel(const float* __restrict__ xyz, const int* __restrict__ nbrs,
                                 const float* __restrict__ dist, _Float16* __restrict__ rpe) {
  const int m = blockIdx.x * 256 + threadIdx.x;
  if (m >= MNK) return;
  const int n = (m / KNN) % NPTS;
  const int b = m / (KNN * NPTS);
  const int nb = nbrs[m];
  const float* pc = xyz + (size_t)(b * NPTS + n) * 3;
  const float* pn = xyz + (size_t)(b * NPTS + nb) * 3;
  const float cx = pc[0], cy = pc[1], cz = pc[2];
  const float nx = pn[0], ny = pn[1], nz = pn[2];
  _Float16* d = rpe + (size_t)m * 32;
  d[0] = (_Float16)cx; d[1] = (_Float16)cy; d[2] = (_Float16)cz;
  d[3] = (_Float16)nx; d[4] = (_Float16)ny; d[5] = (_Float16)nz;
  d[6] = (_Float16)(cx - nx); d[7] = (_Float16)(cy - ny); d[8] = (_Float16)(cz - nz);
  d[9] = (_Float16)dist[m];
#pragma unroll
  for (int c = 10; c < 32; ++c) d[c] = (_Float16)0.0f;
}

// gather per-point features along neighbor index into columns [rowoff, rowoff+Cs)
__global__ void gather_concat_kernel(const _Float16* __restrict__ src,
                                     const int* __restrict__ nbrs,
                                     _Float16* __restrict__ dst,
                                     int Cs, int ldD, int rowoff, int total) {
  const int tid = blockIdx.x * 256 + threadIdx.x;
  if (tid >= total) return;
  const int c = tid % Cs, m = tid / Cs;
  const int b = m / (KNN * NPTS);
  const int nb = nbrs[m];
  dst[(size_t)m * ldD + rowoff + c] = src[(size_t)(b * NPTS + nb) * Cs + c];
}

// softmax over K of scores S (f32, C x MNK), weighted sum of F (f16, MNK x 128)
// -> pooled (f16, MN x C)
__global__ void softmax_pool_kernel(const float* __restrict__ S,
                                    const _Float16* __restrict__ F,
                                    _Float16* __restrict__ pooled, int C) {
  const int tid = blockIdx.x * 256 + threadIdx.x;
  if (tid >= C * MN) return;
  const int c = tid / MN, bn = tid % MN;
  const float* s = S + (size_t)c * MNK + (size_t)bn * KNN;
  float mx = -3.0e38f;
#pragma unroll
  for (int k = 0; k < KNN; ++k) mx = fmaxf(mx, s[k]);
  float e[KNN]; float sum = 0.f;
#pragma unroll
  for (int k = 0; k < KNN; ++k) { e[k] = __expf(s[k] - mx); sum += e[k]; }
  float acc = 0.f;
#pragma unroll
  for (int k = 0; k < KNN; ++k)
    acc += e[k] * (float)F[(size_t)(bn * KNN + k) * 128 + c];
  pooled[(size_t)bn * C + c] = (_Float16)(acc / sum);
}

// out = leaky_relu(A + B, 0.01), laid out (B, 256, N, 1) f32
__global__ void combine_out_kernel(const float* __restrict__ A,
                                   const float* __restrict__ Bc,
                                   float* __restrict__ out) {
  const int tid = blockIdx.x * 256 + threadIdx.x;
  if (tid >= NBAT * 256 * NPTS) return;
  const int n = tid % NPTS;
  const int c = (tid / NPTS) % 256;
  const int b = tid / (256 * NPTS);
  const size_t i = (size_t)c * MN + (size_t)b * NPTS + n;
  const float v = A[i] + Bc[i];
  out[tid] = (v > 0.f) ? v : 0.01f * v;
}

// ---------------------------------------------------------------------------
extern "C" void kernel_launch(void* const* d_in, const int* in_sizes, int n_in,
                              void* d_out, int out_size, void* d_ws, size_t ws_size,
                              hipStream_t stream) {
  (void)in_sizes; (void)n_in; (void)out_size; (void)ws_size;
  const float* xyz   = (const float*)d_in[0];
  const float* feat  = (const float*)d_in[1];
  const float* m1_w  = (const float*)d_in[2];
  const float* m1_b  = (const float*)d_in[3];
  const float* m1_g  = (const float*)d_in[4];
  const float* m1_be = (const float*)d_in[5];
  const float* r1_w  = (const float*)d_in[6];
  const float* r1_b  = (const float*)d_in[7];
  const float* r1_g  = (const float*)d_in[8];
  const float* r1_be = (const float*)d_in[9];
  const float* r2_w  = (const float*)d_in[10];
  const float* r2_b  = (const float*)d_in[11];
  const float* r2_g  = (const float*)d_in[12];
  const float* r2_be = (const float*)d_in[13];
  const float* p1s_w = (const float*)d_in[14];
  const float* p1_w  = (const float*)d_in[15];
  const float* p1_b  = (const float*)d_in[16];
  const float* p1_g  = (const float*)d_in[17];
  const float* p1_be = (const float*)d_in[18];
  const float* p2s_w = (const float*)d_in[19];
  const float* p2_w  = (const float*)d_in[20];
  const float* p2_b  = (const float*)d_in[21];
  const float* p2_g  = (const float*)d_in[22];
  const float* p2_be = (const float*)d_in[23];
  const float* m2_w  = (const float*)d_in[24];
  const float* m2_b  = (const float*)d_in[25];
  const float* m2_g  = (const float*)d_in[26];
  const float* m2_be = (const float*)d_in[27];
  const float* sc_w  = (const float*)d_in[28];
  const float* sc_b  = (const float*)d_in[29];
  const float* sc_g  = (const float*)d_in[30];
  const float* sc_be = (const float*)d_in[31];

  char* ws = (char*)d_ws;
  const size_t MB = (size_t)1 << 20;
  int*      nbrs    = (int*)(ws + 0);            // 1 MB
  float*    dist    = (float*)(ws + 2 * MB);     // 1 MB
  float*    st_mean = (float*)(ws + 4 * MB);     // 1 KB
  float*    st_rstd = st_mean + 256;             // 1 KB
  float*    zbias   = st_mean + 512;             // 1 KB (zero bias vector)
  _Float16* wts     = (_Float16*)(ws + 5 * MB);  // ~0.25 MB
  _Float16* featin  = (_Float16*)(ws + 6 * MB);  // 2 MB  (MN x 64)
  _Float16* feats   = (_Float16*)(ws + 8 * MB);  // 2 MB  (MN x 64)
  _Float16* f1res   = (_Float16*)(ws + 10 * MB); // 2 MB  (MN x 64)
  _Float16* pooled  = (_Float16*)(ws + 12 * MB); // 4 MB  (MN x 128)
  _Float16* g2      = (_Float16*)(ws + 16 * MB); // 4 MB  (MN x 128)
  _Float16* rpe     = (_Float16*)(ws + 20 * MB); // 16 MB (MNK x 32)
  float*    Am2     = (float*)(ws + 36 * MB);    // 16 MB (256 x MN)
  float*    Asc     = (float*)(ws + 52 * MB);    // 16 MB (256 x MN)
  _Float16* bufA    = (_Float16*)(ws + 68 * MB); // 64 MB (MNK x 128)
  _Float16* bufB    = (_Float16*)(ws + 132 * MB);// 64 MB (MNK x 128)
  float*    Yscr    = (float*)(ws + 196 * MB);   // 128 MB (128 x MNK)

  _Float16* w_m1  = wts;
  _Float16* w_r1  = w_m1 + 64 * 64;
  _Float16* w_r2  = w_r1 + 64 * 32;
  _Float16* w_p1s = w_r2 + 64 * 64;
  _Float16* w_p1  = w_p1s + 128 * 128;
  _Float16* w_p2s = w_p1 + 64 * 128;
  _Float16* w_p2  = w_p2s + 128 * 128;
  _Float16* w_m2  = w_p2 + 128 * 128;
  _Float16* w_sc  = w_m2 + 256 * 128;

  zero_f32_kernel<<<1, 256, 0, stream>>>(zbias, 256);

  auto cvt = [&](const float* s, _Float16* d, int Co, int Ci, int Cip) {
    const int tot = Co * Cip;
    cvt_w_kernel<<<(tot + 255) / 256, 256, 0, stream>>>(s, d, Ci, Cip, tot);
  };
  cvt(m1_w,  w_m1,  64, 64, 64);
  cvt(r1_w,  w_r1,  64, 10, 32);
  cvt(r2_w,  w_r2,  64, 64, 64);
  cvt(p1s_w, w_p1s, 128, 128, 128);
  cvt(p1_w,  w_p1,  64, 128, 128);
  cvt(p2s_w, w_p2s, 128, 128, 128);
  cvt(p2_w,  w_p2,  128, 128, 128);
  cvt(m2_w,  w_m2,  256, 128, 128);
  cvt(sc_w,  w_sc,  256, 64, 64);

  cvt_featin_kernel<<<(NBAT * 64 * NPTS) / 256, 256, 0, stream>>>(feat, featin);
  knn_kernel<<<dim3(NPTS / 256, NBAT), 256, 0, stream>>>(xyz, nbrs, dist);
  rpe_build_kernel<<<MNK / 256, 256, 0, stream>>>(xyz, nbrs, dist, rpe);

  // Each wave covers 64 output channels (4 WMMA Cout-tiles) x 16 points.
  auto gemm = [&](const _Float16* Wp, const _Float16* Xp, float* Yp, const float* bias,
                  int Co, int Cip, int M, int ldX) {
    const int tiles = (Co / 64) * (M / 16);
    const int blocks = (tiles + 7) / 8;
    if (Cip == 32)
      gemm_wmma_f16<32, 4><<<blocks, 256, 0, stream>>>(Wp, Xp, Yp, bias, M, ldX, tiles);
    else if (Cip == 64)
      gemm_wmma_f16<64, 4><<<blocks, 256, 0, stream>>>(Wp, Xp, Yp, bias, M, ldX, tiles);
    else
      gemm_wmma_f16<128, 4><<<blocks, 256, 0, stream>>>(Wp, Xp, Yp, bias, M, ldX, tiles);
  };
  auto stats = [&](const float* Y, int Co, int M) {
    stats_kernel<<<Co, 256, 0, stream>>>(Y, M, st_mean, st_rstd);
  };
  auto bnact = [&](const float* Y, const float* g, const float* be, _Float16* d,
                   int Co, int M, int ldD, int rowoff, int act) {
    const int tot = Co * M;
    bn_act_f16_kernel<<<(tot + 255) / 256, 256, 0, stream>>>(
        Y, st_mean, st_rstd, g, be, d, M, ldD, rowoff, act, tot);
  };

  // rpe1 = relu(BN(r1 @ rpe)) -> bufA cols [0,64)
  gemm(w_r1, rpe, Yscr, r1_b, 64, 32, MNK, 32);
  stats(Yscr, 64, MNK);
  bnact(Yscr, r1_g, r1_be, bufA, 64, MNK, 128, 0, 1);

  // feats = leaky0.2(BN(m1 @ feat_in))
  gemm(w_m1, featin, Yscr, m1_b, 64, 64, MN, 64);
  stats(Yscr, 64, MN);
  bnact(Yscr, m1_g, m1_be, feats, 64, MN, 64, 0, 2);

  // concat gathered feats -> bufA cols [64,128)
  gather_concat_kernel<<<(MNK * 64) / 256, 256, 0, stream>>>(
      feats, nbrs, bufA, 64, 128, 64, MNK * 64);

  // attentive pool 1
  gemm(w_p1s, bufA, Yscr, zbias, 128, 128, MNK, 128);
  softmax_pool_kernel<<<(128 * MN) / 256, 256, 0, stream>>>(Yscr, bufA, pooled, 128);
  gemm(w_p1, pooled, Yscr, p1_b, 64, 128, MN, 128);
  stats(Yscr, 64, MN);
  bnact(Yscr, p1_g, p1_be, f1res, 64, MN, 64, 0, 1);

  // rpe2 = relu(BN(r2 @ rpe1)) (rpe1 lives in bufA cols [0,64), ld=128)
  gemm(w_r2, bufA, Yscr, r2_b, 64, 64, MNK, 128);
  stats(Yscr, 64, MNK);
  bnact(Yscr, r2_g, r2_be, bufB, 64, MNK, 128, 0, 1);

  // concat gathered f1res -> bufB cols [64,128)
  gather_concat_kernel<<<(MNK * 64) / 256, 256, 0, stream>>>(
      f1res, nbrs, bufB, 64, 128, 64, MNK * 64);

  // attentive pool 2
  gemm(w_p2s, bufB, Yscr, zbias, 128, 128, MNK, 128);
  softmax_pool_kernel<<<(128 * MN) / 256, 256, 0, stream>>>(Yscr, bufB, pooled, 128);
  gemm(w_p2, pooled, Yscr, p2_b, 128, 128, MN, 128);
  stats(Yscr, 128, MN);
  bnact(Yscr, p2_g, p2_be, g2, 128, MN, 128, 0, 1);

  // out = leaky0.01( BN(m2 @ g2) + BN(sc @ feat_in) )
  gemm(w_m2, g2, Am2, m2_b, 256, 128, MN, 128);
  stats(Am2, 256, MN);
  bn_f32_kernel<<<(256 * MN) / 256, 256, 0, stream>>>(
      Am2, st_mean, st_rstd, m2_g, m2_be, MN, 256 * MN);

  gemm(w_sc, featin, Asc, sc_b, 256, 64, MN, 64);
  stats(Asc, 256, MN);
  bn_f32_kernel<<<(256 * MN) / 256, 256, 0, stream>>>(
      Asc, st_mean, st_rstd, sc_g, sc_be, MN, 256 * MN);

  combine_out_kernel<<<(NBAT * 256 * NPTS) / 256, 256, 0, stream>>>(
      Am2, Asc, (float*)d_out);
}
